// Decoder_26508538151537
// MI455X (gfx1250) — matrix operand order repaired and checked by reading, not compiled, not run on hardware
//
#include <hip/hip_runtime.h>

typedef __attribute__((ext_vector_type(16))) _Float16 v16h;
typedef __attribute__((ext_vector_type(8)))  _Float16 v8h;
typedef __attribute__((ext_vector_type(8)))  float    v8f;

#define B_     64
#define LAT    256
#define ST     32
#define NNODE  256
#define E_     32
#define H_     64
#define NE     (NNODE * E_)   // 8192

// ---------------------------------------------------------------------------
// ws layout:
//   [0,       16384)          h   f32 [64][64]
//   [16384,   16384+1048576)  emb f16 [64][256][32]
// ---------------------------------------------------------------------------

__global__ void k_hidden(const float* __restrict__ z, const float* __restrict__ st,
                         const float* __restrict__ W1, const float* __restrict__ b1,
                         float* __restrict__ h) {
  const int b = blockIdx.x;      // 64 blocks
  const int n = threadIdx.x;     // 64 threads
  float acc = b1[n];
  const float* zb = z + b * LAT;
  for (int k = 0; k < LAT; ++k) acc += zb[k] * W1[k * H_ + n];
  const float* sb = st + b * ST;
  for (int k = 0; k < ST; ++k)  acc += sb[k] * W1[(LAT + k) * H_ + n];
  h[b * H_ + n] = fmaxf(acc, 0.f);
}

__global__ void k_nodeemb(const float* __restrict__ h, const float* __restrict__ W2,
                          const float* __restrict__ b2, _Float16* __restrict__ emb) {
  const int idx = blockIdx.x * blockDim.x + threadIdx.x;  // 524288 total
  const int b = idx >> 13;
  const int c = idx & (NE - 1);
  const float* hb = h + b * H_;
  float acc = b2[c];
#pragma unroll 8
  for (int k = 0; k < H_; ++k) acc += hb[k] * W2[k * NE + c];
  emb[idx] = (_Float16)acc;
}

__device__ __forceinline__ v16h cat16(v8h lo, v8h hiv) {
  v16h r;
#pragma unroll
  for (int t = 0; t < 8; ++t) { r[t] = lo[t]; r[t + 8] = hiv[t]; }
  return r;
}

// Single-VALU-op ReLU: med3(x, 0, +big) clamps to [0, big].
__device__ __forceinline__ float relu1(float x) {
  return __builtin_amdgcn_fmed3f(x, 0.f, 3.0e38f);
}

// One block per (batch, row-pair {r, 254-r}). 4 waves, each owns 16-edge tiles.
// Pairing rows r and 254-r gives every block exactly 256 edges (uniform load).
__launch_bounds__(128)
__global__ void k_edges(const _Float16* __restrict__ emb,
                        const float* __restrict__ We1, const float* __restrict__ be1,
                        const float* __restrict__ We2, const float* __restrict__ be2,
                        const float* __restrict__ We3, const float* __restrict__ be3,
                        float* __restrict__ adj) {
  __shared__ _Float16 emb_s[NNODE * E_];   // 16 KB : batch node embeddings
  __shared__ _Float16 w1_s[H_ * H_];       //  8 KB : We1 transposed [n][k] f16
  __shared__ _Float16 w2_s[H_ * H_];       //  8 KB : We2 transposed [n][k] f16
  __shared__ _Float16 esc[4 * 16 * H_];    //  8 KB : per-wave activation bounce
  __shared__ float    red[4 * 16 * 16];    //  4 KB : per-wave final reduction

  const int blk  = blockIdx.x;
  const int b    = blk >> 7;        // batch
  const int r    = blk & 127;       // row-pair selector
  const int tid  = threadIdx.x;
  const int wave = tid >> 5;
  const int lane = tid & 31;
  const int ln   = lane & 15;       // N index (B/C/D frags), M index (A frags)
  const int hi   = lane >> 4;       // half-wave selector

  // Stage this batch's node embeddings into LDS (16 KB = 1024 uint4).
  {
    const uint4* src = (const uint4*)(emb + (size_t)b * NNODE * E_);
    uint4* dst = (uint4*)emb_s;
    for (int t = tid; t < 1024; t += 128) dst[t] = src[t];
  }
  // Stage transposed f16 weights: w_s[n][k] = (f16)W[k][n].
  for (int t = tid; t < H_ * H_; t += 128) {
    const int n = t >> 6, k = t & 63;
    w1_s[n * H_ + k] = (_Float16)We1[k * H_ + n];
    w2_s[n * H_ + k] = (_Float16)We2[k * H_ + n];
  }
  __syncthreads();

  // Per-lane layout constants (contiguous K order k = 16*hi + s for both A and B, wave32).
  const int kb  = hi ? 16 : 0;   // A frag: halves 0..7  hold K = kb..kb+7
  const int kb2 = kb + 8;        //         halves 8..15 hold K = kb2..kb2+7
  const int wkb = hi ? 16 : 0;   // B frag: halves 0..15 hold K = wkb..wkb+15

  // Persistent weight (B) fragments: layer-1 K-chunk1 (zj) and both of layer-2.
  v16h W1f1[4], W2f[2][4];
#pragma unroll
  for (int nt = 0; nt < 4; ++nt) {
    const int n  = nt * 16 + ln;
    const int k1 = 32 + wkb;
    W1f1[nt]   = cat16(*(const v8h*)&w1_s[n * H_ + k1],
                       *(const v8h*)&w1_s[n * H_ + k1 + 8]);
    W2f[0][nt] = cat16(*(const v8h*)&w2_s[n * H_ + wkb],
                       *(const v8h*)&w2_s[n * H_ + wkb + 8]);
    W2f[1][nt] = cat16(*(const v8h*)&w2_s[n * H_ + 32 + wkb],
                       *(const v8h*)&w2_s[n * H_ + 32 + wkb + 8]);
  }

  // Biases / output weights, indexed by this lane's N.
  _Float16 bias1[4];
  float bias2[4], w3[4];
#pragma unroll
  for (int nt = 0; nt < 4; ++nt) {
    bias1[nt] = (_Float16)be1[nt * 16 + ln];
    bias2[nt] = be2[nt * 16 + ln];
    w3[nt]    = We3[nt * 16 + ln];
  }
  const float be3s = be3[0];

  _Float16* myesc = &esc[wave * 16 * H_];
  float*    myred = &red[wave * 16 * 16];
  const size_t obase = (size_t)b * NNODE * NNODE;

  for (int pass = 0; pass < 2; ++pass) {
    const int i = pass ? (NNODE - 2 - r) : r;
    if (pass && i == r) break;                 // r == 127: single middle row

    // Precompute per-row layer-1 partial: ci0 = bias1 + zi @ We1[0:32,:].
    // zi = emb[i] is identical for every edge of this row, so this WMMA
    // happens once per row instead of once per tile (saves 25% of WMMAs).
    // f16 accumulator: layer-1 output is quantized to f16 anyway.
    v8h ci0[4];
    {
      v16h a0 = cat16(*(const v8h*)&emb_s[i * E_ + kb],
                      *(const v8h*)&emb_s[i * E_ + kb2]);
#pragma unroll
      for (int nt = 0; nt < 4; ++nt) {
        const int n = nt * 16 + ln;
        v16h w0 = cat16(*(const v8h*)&w1_s[n * H_ + wkb],
                        *(const v8h*)&w1_s[n * H_ + wkb + 8]);
        v8h c;
#pragma unroll
        for (int q = 0; q < 8; ++q) c[q] = bias1[nt];
        ci0[nt] = __builtin_amdgcn_wmma_f16_16x16x32_f16(false, a0, false, w0,
                                                         (short)0, c, false, false);
      }
    }

    const int cnt    = (NNODE - 1) - i;
    const int ntiles = (cnt + 15) >> 4;

    for (int t = wave; t < ntiles; t += 4) {
      const int jbase = i + 1 + t * 16;
      const int j  = jbase + ln;
      const int jc = j < NNODE ? j : (NNODE - 1);
      v16h a1 = cat16(*(const v8h*)&emb_s[jc * E_ + kb],
                      *(const v8h*)&emb_s[jc * E_ + kb2]);

      // Layer 1 (zj half only; zi half folded into ci0), f16 D-matrix.
      // 16-bit D layout: element q of the v8h maps to m = q + 8*hi, n = ln.
#pragma unroll
      for (int nt = 0; nt < 4; ++nt) {
        v8h d = __builtin_amdgcn_wmma_f16_16x16x32_f16(false, a1, false, W1f1[nt],
                                                       (short)0, ci0[nt],
                                                       false, false);
#pragma unroll
        for (int q = 0; q < 8; ++q) {
          _Float16 v = d[q];
          v = (v > (_Float16)0) ? v : (_Float16)0;
          myesc[(q + hi * 8) * H_ + nt * 16 + ln] = v;
        }
      }

      // Layer-2 A fragments (LDS is in-order within a wave).
      v16h e0 = cat16(*(const v8h*)&myesc[ln * H_ + kb],
                      *(const v8h*)&myesc[ln * H_ + kb2]);
      v16h e1 = cat16(*(const v8h*)&myesc[ln * H_ + 32 + kb],
                      *(const v8h*)&myesc[ln * H_ + 32 + kb2]);

      // Layer 2 (f32 accumulate) + fused layer-3 partial dot with We3.
      float part[8];
#pragma unroll
      for (int q = 0; q < 8; ++q) part[q] = 0.f;
#pragma unroll
      for (int nt = 0; nt < 4; ++nt) {
        v8f c;
#pragma unroll
        for (int q = 0; q < 8; ++q) c[q] = bias2[nt];
        c = __builtin_amdgcn_wmma_f32_16x16x32_f16(false, e0, false, W2f[0][nt],
                                                   (short)0, c, false, false);
        c = __builtin_amdgcn_wmma_f32_16x16x32_f16(false, e1, false, W2f[1][nt],
                                                   (short)0, c, false, false);
#pragma unroll
        for (int q = 0; q < 8; ++q) part[q] += relu1(c[q]) * w3[nt];
      }
#pragma unroll
      for (int q = 0; q < 8; ++q) myred[(q + hi * 8) * 16 + ln] = part[q];

      // Cross-lane reduction over the 16 N-lanes; lanes 0..15 own edges m=ln.
      if (hi == 0) {
        const float4* pr = (const float4*)&myred[ln * 16];
        const float4 q0 = pr[0], q1 = pr[1], q2 = pr[2], q3 = pr[3];
        float s = be3s
                + (((q0.x + q0.y) + (q0.z + q0.w)) +
                   ((q1.x + q1.y) + (q1.z + q1.w)))
                + (((q2.x + q2.y) + (q2.z + q2.w)) +
                   ((q3.x + q3.y) + (q3.z + q3.w)));
        const int jj = jbase + ln;
        if (jj < NNODE) {
          adj[obase + (size_t)i * NNODE + jj] = s;
          adj[obase + (size_t)jj * NNODE + i] = s;
        }
      }
    }
  }
}

__global__ void k_diag(float* __restrict__ adj) {
  const int idx = blockIdx.x * blockDim.x + threadIdx.x;  // 64*256
  const int b = idx >> 8, d = idx & 255;
  adj[(size_t)b * NNODE * NNODE + (size_t)d * (NNODE + 1)] = 0.f;
}

extern "C" void kernel_launch(void* const* d_in, const int* in_sizes, int n_in,
                              void* d_out, int out_size, void* d_ws, size_t ws_size,
                              hipStream_t stream) {
  (void)in_sizes; (void)n_in; (void)out_size; (void)ws_size;
  const float* latent = (const float*)d_in[0];
  const float* stats  = (const float*)d_in[1];
  const float* W1  = (const float*)d_in[2];
  const float* b1  = (const float*)d_in[3];
  const float* W2  = (const float*)d_in[4];
  const float* b2  = (const float*)d_in[5];
  const float* We1 = (const float*)d_in[6];
  const float* be1 = (const float*)d_in[7];
  const float* We2 = (const float*)d_in[8];
  const float* be2 = (const float*)d_in[9];
  const float* We3 = (const float*)d_in[10];
  const float* be3 = (const float*)d_in[11];

  float*     adj = (float*)d_out;
  float*     h   = (float*)d_ws;
  _Float16*  emb = (_Float16*)((char*)d_ws + 16384);

  k_hidden <<<B_, H_, 0, stream>>>(latent, stats, W1, b1, h);
  k_nodeemb<<<(B_ * NE) / 256, 256, 0, stream>>>(h, W2, b2, emb);
  k_edges  <<<B_ * 128, 128, 0, stream>>>(emb, We1, be1, We2, be2,
                                          We3, be3, adj);
  k_diag   <<<B_, NNODE, 0, stream>>>(adj);
}
